// CosmicMultiHeadAttention_20864951124153
// MI455X (gfx1250) — matrix operand order repaired and checked by reading, not compiled, not run on hardware
//
#include <hip/hip_runtime.h>
#include <hip/hip_bf16.h>

#define B_   2
#define S_   4096
#define E_   4096
#define HQ_  32
#define HKV_ 8
#define D_   128
#define BS_  128
#define NB_  (S_ / BS_)
#define GQ_  (HQ_ / HKV_)

typedef __attribute__((ext_vector_type(16))) __bf16     v16bf;
typedef __attribute__((ext_vector_type(8)))  float      v8f;
typedef __attribute__((ext_vector_type(4)))  unsigned int u32x4;
typedef int v4i_ __attribute__((vector_size(16)));   // matches builtin pointee type

union ABFrag { u32x4 u[2]; v16bf bf; };

#if __has_builtin(__builtin_amdgcn_global_load_async_to_lds_b128)
#define HAVE_ASYNC_LDS 1
#else
#define HAVE_ASYNC_LDS 0
#endif

__device__ __forceinline__ void wait_asynccnt(int n) {
#if HAVE_ASYNC_LDS
#if __has_builtin(__builtin_amdgcn_s_wait_asynccnt)
  if (n == 0) __builtin_amdgcn_s_wait_asynccnt(0);
  else        __builtin_amdgcn_s_wait_asynccnt(8);
#else
  if (n == 0) asm volatile("s_wait_asynccnt 0x0" ::: "memory");
  else        asm volatile("s_wait_asynccnt 0x8" ::: "memory");
#endif
#else
  (void)n;
#endif
}

// 16-byte global -> LDS copy. Async (ASYNCcnt) on CDNA5 when available.
__device__ __forceinline__ void copy16_g2l(const void* g, void* lds) {
#if HAVE_ASYNC_LDS
  __builtin_amdgcn_global_load_async_to_lds_b128(
      (__attribute__((address_space(1))) v4i_*)(unsigned long long)g,
      (__attribute__((address_space(3))) v4i_*)(unsigned)(unsigned long long)lds,
      0, 0);
#else
  *(u32x4*)lds = *(const u32x4*)g;
#endif
}

__device__ __forceinline__ unsigned short f2bf(float f) {
  union { float f; unsigned int u; } v; v.f = f;
  unsigned int r = v.u + 0x7fffu + ((v.u >> 16) & 1u);
  return (unsigned short)(r >> 16);
}
__device__ __forceinline__ float bf2f(unsigned short h) {
  union { unsigned int u; float f; } v; v.u = ((unsigned int)h) << 16;
  return v.f;
}

// ---------------------------------------------------------------- convert X
__global__ void f32_to_bf16_kernel(const float* __restrict__ src,
                                   unsigned short* __restrict__ dst, int n) {
  int i = blockIdx.x * blockDim.x + threadIdx.x;
  if (i < n) dst[i] = f2bf(src[i]);
}

// ------------------------------------------- transpose W (K,N) -> bf16 (N,K)
__global__ void transpose_to_bf16_kernel(const float* __restrict__ w,
                                         unsigned short* __restrict__ wt,
                                         int K, int N) {
  __shared__ float tile[32][33];
  int tx = threadIdx.x & 31, ty = threadIdx.x >> 5;   // 32 x 8
  int kb = blockIdx.y * 32, nb = blockIdx.x * 32;
#pragma unroll
  for (int i = 0; i < 32; i += 8)
    tile[ty + i][tx] = w[(size_t)(kb + ty + i) * N + (nb + tx)];
  __syncthreads();
#pragma unroll
  for (int i = 0; i < 32; i += 8)
    wt[(size_t)(nb + ty + i) * K + (kb + tx)] = f2bf(tile[tx][ty + i]);
}

// ------------------------------------------------------------ bf16 WMMA GEMM
// C(MxN) = A(MxK) * Bt(NxK)^T + bias ; A,Bt bf16 row-major, accum fp32.
// Double-buffered LDS, async global->LDS prefetch of the next K-chunk.
#define GTM 128
#define GTN 128
#define GKC 64
#define GLD (GKC + 8)   // 72 halves = 144B row stride, 16B aligned

__global__ __launch_bounds__(256)
void gemm_bf16_kernel(const unsigned short* __restrict__ A,
                      const unsigned short* __restrict__ Bt,
                      const float* __restrict__ bias,
                      void* __restrict__ Cout, int write_f32,
                      int M, int N, int K) {
  __shared__ unsigned short lA[2][GTM][GLD];
  __shared__ unsigned short lB[2][GTN][GLD];
  int tid  = threadIdx.x;
  int wave = tid >> 5, lane = tid & 31;
  int l = lane & 15, hi = lane >> 4;
  int wm = wave >> 2, wn = wave & 3;          // 2 (M) x 4 (N) wave grid
  size_t tM = (size_t)blockIdx.y * GTM;
  size_t tN = (size_t)blockIdx.x * GTN;

  const v8f vz = {0.f, 0.f, 0.f, 0.f, 0.f, 0.f, 0.f, 0.f};
  v8f acc[4][2];
#pragma unroll
  for (int i = 0; i < 4; ++i)
#pragma unroll
    for (int j = 0; j < 2; ++j) acc[i][j] = vz;

  // each thread owns 4 A-chunks + 4 B-chunks of 16B per tile stage
  auto issue_tile = [&](int buf, int k0) {
#pragma unroll
    for (int i = 0; i < 4; ++i) {
      int g = tid + 256 * i;                   // 0..1023
      int row = g >> 3, c = g & 7;
      copy16_g2l(&A[(tM + row) * K + k0 + c * 8], &lA[buf][row][c * 8]);
      copy16_g2l(&Bt[(tN + row) * K + k0 + c * 8], &lB[buf][row][c * 8]);
    }
  };

  int nIter = K / GKC;
  issue_tile(0, 0);                            // prologue: tile 0 in flight
  for (int it = 0; it < nIter; ++it) {
    int buf = it & 1;
    __syncthreads();                           // everyone done reading buf^1
    if (it + 1 < nIter) {
      issue_tile(buf ^ 1, (it + 1) * GKC);     // prefetch next tile
      wait_asynccnt(8);                        // my current-tile loads retired
    } else {
      wait_asynccnt(0);
    }
    __syncthreads();                           // current tile visible to all
#pragma unroll
    for (int kk = 0; kk < GKC; kk += 32) {
      ABFrag af[4], bfr[2];
#pragma unroll
      for (int i = 0; i < 4; ++i) {
        const unsigned short* p = &lA[buf][wm * 64 + i * 16 + l][kk + hi * 8];
        af[i].u[0] = *(const u32x4*)p;          // K 0-7  / 8-15
        af[i].u[1] = *(const u32x4*)(p + 16);   // K 16-23/ 24-31
      }
#pragma unroll
      for (int j = 0; j < 2; ++j) {
        const unsigned short* p = &lB[buf][wn * 32 + j * 16 + l][kk + hi * 8];
        bfr[j].u[0] = *(const u32x4*)p;         // K 0-7  / 8-15
        bfr[j].u[1] = *(const u32x4*)(p + 16);
      }
#pragma unroll
      for (int i = 0; i < 4; ++i)
#pragma unroll
        for (int j = 0; j < 2; ++j)
          acc[i][j] = __builtin_amdgcn_wmma_f32_16x16x32_bf16(
              false, af[i].bf, false, bfr[j].bf, (short)0, acc[i][j],
              false, false);
    }
  }
#pragma unroll
  for (int i = 0; i < 4; ++i)
#pragma unroll
    for (int j = 0; j < 2; ++j)
#pragma unroll
      for (int r = 0; r < 8; ++r) {
        size_t row = tM + wm * 64 + i * 16 + hi * 8 + r;
        size_t col = tN + wn * 32 + j * 16 + l;
        float v = acc[i][j][r] + bias[col];
        if (write_f32) ((float*)Cout)[row * N + col] = v;
        else ((unsigned short*)Cout)[row * N + col] = f2bf(v);
      }
}

// ------------------------------------------------------------------- RoPE
// src: (B*S, H*D) bf16 ; dst: (B, H, S, D) bf16 ; one thread per (pair)
__global__ void rope_kernel(const unsigned short* __restrict__ src,
                            unsigned short* __restrict__ dst, int H) {
  int gid = blockIdx.x * blockDim.x + threadIdx.x;   // ((b*H+h)*S+s)*64+d
  int d = gid & 63;
  int t = gid >> 6;
  int s = t & (S_ - 1);
  int bh = t >> 12;
  int h = bh % H;
  int b = bh / H;
  size_t srow = (size_t)(b * S_ + s) * ((size_t)H * D_) + (size_t)h * D_;
  float x1 = bf2f(src[srow + d]);
  float x2 = bf2f(src[srow + d + 64]);
  float freq = __expf(-(float)d * (9.210340371976184f / 64.0f)); // 10000^(-d/64)
  float ang = (float)s * freq, sn, cs;
  __sincosf(ang, &sn, &cs);
  size_t drow = ((size_t)(b * H + h) * S_ + s) * D_;
  dst[drow + d]      = f2bf(x1 * cs - x2 * sn);
  dst[drow + d + 64] = f2bf(x2 * cs + x1 * sn);
}

// -------------------- V repack: (B*S, HKV*D) -> (B, HKV, NB, D, BS) bf16
__global__ void v_repack_kernel(const unsigned short* __restrict__ vf,
                                unsigned short* __restrict__ vt) {
  int gid = blockIdx.x * blockDim.x + threadIdx.x;
  int key = gid & 127;
  int d = (gid >> 7) & 127;
  int rest = gid >> 14;           // b*HKV*NB
  int nb = rest & (NB_ - 1);
  int bh = rest >> 5;             // b*HKV + h
  int b = bh / HKV_, h = bh % HKV_;
  vt[gid] = vf[(size_t)(b * S_ + nb * BS_ + key) * (HKV_ * D_) + h * D_ + d];
}

// --------------------------------------------------- block-local attention
__global__ __launch_bounds__(256)
void attn_kernel(const unsigned short* __restrict__ qr,   // (B,HQ,S,D)
                 const unsigned short* __restrict__ kr,   // (B,HKV,S,D)
                 const unsigned short* __restrict__ vt,   // (B,HKV,NB,D,BS)
                 const unsigned char* __restrict__ mask,  // (B,S)
                 unsigned short* __restrict__ obf) {      // (B*S, HQ*D)
  __shared__ unsigned short lQ[128][136];   // also reused for probs (per-wave rows)
  __shared__ unsigned short lK[128][136];
  __shared__ unsigned short lV[128][136];
  int gid = blockIdx.x;
  int h = gid % HQ_;
  int n = (gid / HQ_) % NB_;
  int b = gid / (HQ_ * NB_);
  int hk = h / GQ_;
  const unsigned short* qsrc = qr + ((size_t)(b * HQ_ + h) * S_ + n * BS_) * D_;
  const unsigned short* ksrc = kr + ((size_t)(b * HKV_ + hk) * S_ + n * BS_) * D_;
  const unsigned short* vsrc = vt + ((size_t)((b * HKV_ + hk) * NB_ + n)) * (D_ * BS_);
  int tid = threadIdx.x;
#pragma unroll
  for (int i = 0; i < 8; ++i) {              // 2048 x 16B chunks per tile
    int g = tid + 256 * i;
    int row = g >> 4, c = g & 15;
    copy16_g2l(&qsrc[row * 128 + c * 8], &lQ[row][c * 8]);
    copy16_g2l(&ksrc[row * 128 + c * 8], &lK[row][c * 8]);
    copy16_g2l(&vsrc[row * 128 + c * 8], &lV[row][c * 8]);
  }
  wait_asynccnt(0);
  __syncthreads();

  int wave = tid >> 5, lane = tid & 31;
  int l = lane & 15, hi = lane >> 4;
  int m0 = wave * 16;                        // 16 q-rows per wave

  bool mk[8];
#pragma unroll
  for (int nf = 0; nf < 8; ++nf)
    mk[nf] = mask[(size_t)b * S_ + n * BS_ + nf * 16 + l] != 0;

  const v8f vz = {0.f, 0.f, 0.f, 0.f, 0.f, 0.f, 0.f, 0.f};
  v8f sc[8];
#pragma unroll
  for (int nf = 0; nf < 8; ++nf) sc[nf] = vz;

  // ---- stage 1: S = Q * K^T
#pragma unroll
  for (int kk = 0; kk < 128; kk += 32) {
    ABFrag a;
    const unsigned short* pa = &lQ[m0 + l][kk + hi * 8];
    a.u[0] = *(const u32x4*)pa;
    a.u[1] = *(const u32x4*)(pa + 16);
#pragma unroll
    for (int nf = 0; nf < 8; ++nf) {
      ABFrag bb;
      const unsigned short* pb = &lK[nf * 16 + l][kk + hi * 8];
      bb.u[0] = *(const u32x4*)pb;
      bb.u[1] = *(const u32x4*)(pb + 16);
      sc[nf] = __builtin_amdgcn_wmma_f32_16x16x32_bf16(
          false, a.bf, false, bb.bf, (short)0, sc[nf], false, false);
    }
  }

  const float scale = 0.08838834764831845f;  // 1/sqrt(128)
#pragma unroll
  for (int nf = 0; nf < 8; ++nf)
#pragma unroll
    for (int r = 0; r < 8; ++r)
      sc[nf][r] = mk[nf] ? sc[nf][r] * scale : -1e30f;

  // ---- softmax over 128 keys (8 frags x 16 lanes per half-wave)
#pragma unroll
  for (int r = 0; r < 8; ++r) {
    float mx = -3.0e38f;
#pragma unroll
    for (int nf = 0; nf < 8; ++nf) mx = fmaxf(mx, sc[nf][r]);
#pragma unroll
    for (int o = 8; o >= 1; o >>= 1) mx = fmaxf(mx, __shfl_xor(mx, o, 32));
    float sum = 0.f;
#pragma unroll
    for (int nf = 0; nf < 8; ++nf) {
      float e = __expf(sc[nf][r] - mx);
      sc[nf][r] = e;
      sum += e;
    }
#pragma unroll
    for (int o = 8; o >= 1; o >>= 1) sum += __shfl_xor(sum, o, 32);
    float inv = 1.0f / sum;
    int rowm = m0 + hi * 8 + r;              // this wave's own rows only
#pragma unroll
    for (int nf = 0; nf < 8; ++nf)
      lQ[rowm][nf * 16 + l] = f2bf(sc[nf][r] * inv);
  }
  // no barrier: each wave reads back only the lQ rows it just wrote

  // ---- stage 2: O = P * V   (Vt is (d, key) => Bt operand)
  v8f oc[8];
#pragma unroll
  for (int nd = 0; nd < 8; ++nd) oc[nd] = vz;
#pragma unroll
  for (int kk = 0; kk < 128; kk += 32) {
    ABFrag a;
    const unsigned short* pa = &lQ[m0 + l][kk + hi * 8];
    a.u[0] = *(const u32x4*)pa;
    a.u[1] = *(const u32x4*)(pa + 16);
#pragma unroll
    for (int nd = 0; nd < 8; ++nd) {
      ABFrag bb;
      const unsigned short* pb = &lV[nd * 16 + l][kk + hi * 8];
      bb.u[0] = *(const u32x4*)pb;
      bb.u[1] = *(const u32x4*)(pb + 16);
      oc[nd] = __builtin_amdgcn_wmma_f32_16x16x32_bf16(
          false, a.bf, false, bb.bf, (short)0, oc[nd], false, false);
    }
  }
#pragma unroll
  for (int nd = 0; nd < 8; ++nd)
#pragma unroll
    for (int r = 0; r < 8; ++r) {
      int m = m0 + hi * 8 + r;
      size_t srow = (size_t)(b * S_ + n * BS_ + m);
      obf[srow * (HQ_ * D_) + h * D_ + nd * 16 + l] = f2bf(oc[nd][r]);
    }
}

// ---------------------------------------------------------------- launcher
extern "C" void kernel_launch(void* const* d_in, const int* in_sizes, int n_in,
                              void* d_out, int out_size, void* d_ws, size_t ws_size,
                              hipStream_t stream) {
  const float* x  = (const float*)d_in[0];
  const float* wq = (const float*)d_in[1];
  const float* bq = (const float*)d_in[2];
  const float* wk = (const float*)d_in[3];
  const float* bk = (const float*)d_in[4];
  const float* wv = (const float*)d_in[5];
  const float* bv = (const float*)d_in[6];
  const float* wo = (const float*)d_in[7];
  const float* bo = (const float*)d_in[8];
  const unsigned char* mask = (const unsigned char*)d_in[9];

  char* ws = (char*)d_ws;
  size_t off = 0;
  auto alloc = [&](size_t bytes) -> char* {
    char* p = ws + off;
    off += (bytes + 255) & ~(size_t)255;
    return p;
  };
  const size_t BS_E = (size_t)B_ * S_;
  unsigned short* xb  = (unsigned short*)alloc(BS_E * E_ * 2);
  unsigned short* wqt = (unsigned short*)alloc((size_t)(HQ_ * D_) * E_ * 2);
  unsigned short* wkt = (unsigned short*)alloc((size_t)(HKV_ * D_) * E_ * 2);
  unsigned short* wvt = (unsigned short*)alloc((size_t)(HKV_ * D_) * E_ * 2);
  unsigned short* wot = (unsigned short*)alloc((size_t)E_ * (HQ_ * D_) * 2);
  unsigned short* qf  = (unsigned short*)alloc(BS_E * (HQ_ * D_) * 2);
  unsigned short* kf  = (unsigned short*)alloc(BS_E * (HKV_ * D_) * 2);
  unsigned short* vf  = (unsigned short*)alloc(BS_E * (HKV_ * D_) * 2);
  unsigned short* qr  = (unsigned short*)alloc(BS_E * (HQ_ * D_) * 2);
  unsigned short* kr  = (unsigned short*)alloc(BS_E * (HKV_ * D_) * 2);
  unsigned short* vtb = (unsigned short*)alloc(BS_E * (HKV_ * D_) * 2);
  unsigned short* obf = qf;   // Q projection buffer is dead after RoPE; reuse

  // 1) X -> bf16
  {
    int n = (int)(BS_E * E_);
    f32_to_bf16_kernel<<<n / 256, 256, 0, stream>>>(x, xb, n);
  }
  // 2) weight transposes -> bf16 (N,K)
  transpose_to_bf16_kernel<<<dim3((HQ_ * D_) / 32, E_ / 32), 256, 0, stream>>>(wq, wqt, E_, HQ_ * D_);
  transpose_to_bf16_kernel<<<dim3((HKV_ * D_) / 32, E_ / 32), 256, 0, stream>>>(wk, wkt, E_, HKV_ * D_);
  transpose_to_bf16_kernel<<<dim3((HKV_ * D_) / 32, E_ / 32), 256, 0, stream>>>(wv, wvt, E_, HKV_ * D_);
  transpose_to_bf16_kernel<<<dim3(E_ / 32, (HQ_ * D_) / 32), 256, 0, stream>>>(wo, wot, HQ_ * D_, E_);
  // 3) QKV projections (bf16 out)
  gemm_bf16_kernel<<<dim3((HQ_ * D_) / GTN, (B_ * S_) / GTM), 256, 0, stream>>>(
      xb, wqt, bq, qf, 0, B_ * S_, HQ_ * D_, E_);
  gemm_bf16_kernel<<<dim3((HKV_ * D_) / GTN, (B_ * S_) / GTM), 256, 0, stream>>>(
      xb, wkt, bk, kf, 0, B_ * S_, HKV_ * D_, E_);
  gemm_bf16_kernel<<<dim3((HKV_ * D_) / GTN, (B_ * S_) / GTM), 256, 0, stream>>>(
      xb, wvt, bv, vf, 0, B_ * S_, HKV_ * D_, E_);
  // 4) RoPE + V repack
  rope_kernel<<<(B_ * HQ_ * S_ * 64) / 256, 256, 0, stream>>>(qf, qr, HQ_);
  rope_kernel<<<(B_ * HKV_ * S_ * 64) / 256, 256, 0, stream>>>(kf, kr, HKV_);
  v_repack_kernel<<<(B_ * HKV_ * NB_ * D_ * BS_) / 256, 256, 0, stream>>>(vf, vtb);
  // 5) block-local attention (writes obf, aliasing dead qf)
  attn_kernel<<<B_ * NB_ * HQ_, 256, 0, stream>>>(qr, kr, vtb, mask, obf);
  // 6) output projection -> fp32 d_out
  gemm_bf16_kernel<<<dim3(E_ / GTN, (B_ * S_) / GTM), 256, 0, stream>>>(
      obf, wot, bo, d_out, 1, B_ * S_, E_, HQ_ * D_);
  (void)in_sizes; (void)n_in; (void)out_size; (void)ws_size;
}
